// CABlock_26912265077025
// MI455X (gfx1250) — matrix operand order repaired and checked by reading, not compiled, not run on hardware
//
#include <hip/hip_runtime.h>
#include <hip/hip_fp16.h>

typedef _Float16 v16h __attribute__((ext_vector_type(16)));
typedef _Float16 v8h  __attribute__((ext_vector_type(8)));
typedef float    v8f  __attribute__((ext_vector_type(8)));

#define BATCH 4096
#define CCH 3
#define DD 10
#define HH 10
#define WW 40
#define SPAT 12000   // 3*10*10*40
#define N1 4096
#define N2 512

// gfx1250 async global->LDS copy: IOFFSET applies to BOTH lds and global addr
#define ASYNC_B128(ldsoff, gptr, off)                                        \
    asm volatile("global_load_async_to_lds_b128 %0, %1, off offset:" #off    \
                 :: "v"(ldsoff), "v"(gptr) : "memory")

#define WAIT_ASYNCCNT(n) asm volatile("s_wait_asynccnt " #n ::: "memory")

// ---------------------------------------------------------------------------
// fp32 -> fp16 weight conversion (grid-stride)
// ---------------------------------------------------------------------------
__global__ void f32_to_f16(const float* __restrict__ src,
                           _Float16* __restrict__ dst, int n) {
    int i = blockIdx.x * blockDim.x + threadIdx.x;
    int stride = gridDim.x * blockDim.x;
    for (; i < n; i += stride) dst[i] = (_Float16)src[i];
}

// ---------------------------------------------------------------------------
// Gating kernel: one block per batch element.
// Computes sH[10] (shared by gd and gh) and sW[40], then z = x*gd*gh*gw (f16)
// ---------------------------------------------------------------------------
__global__ __launch_bounds__(256)
void gate_kernel(const float* __restrict__ x,
                 const float* __restrict__ conv_w,   // (1,3)
                 const float* __restrict__ F_w,      // (3,1)
                 _Float16* __restrict__ zg)
{
    __shared__ float sumH[CCH * HH];   // mean over (d,w) per (c,h)
    __shared__ float sumW[CCH * WW];   // mean over (d,h) per (c,w)
    __shared__ float sH[HH];
    __shared__ float sW[WW];
    __shared__ float cw[CCH], fw[CCH];

    const int b = blockIdx.x;
    const int tid = threadIdx.x;

    if (tid < CCH * HH) sumH[tid] = 0.f;
    if (tid >= 128 && tid < 128 + CCH * WW) sumW[tid - 128] = 0.f;
    if (tid < CCH) { cw[tid] = conv_w[tid]; fw[tid] = F_w[tid]; }
    __syncthreads();

    const float* xb = x + (size_t)b * SPAT;
    const float invHW = 1.0f / (float)(DD * WW);   // for mean over d,w
    const float invDH = 1.0f / (float)(DD * HH);   // for mean over d,h
    for (int t = tid; t < SPAT; t += 256) {
        float v = xb[t];
        int w = t % WW;
        int h = (t / WW) % HH;
        int c = t / (DD * HH * WW);
        atomicAdd(&sumH[c * HH + h], v * invHW);
        atomicAdd(&sumW[c * WW + w], v * invDH);
    }
    __syncthreads();

    if (tid < HH) {
        float a = 0.f;
        for (int c = 0; c < CCH; ++c) a += cw[c] * sumH[c * HH + tid];
        sH[tid] = fmaxf(a, 0.f);
    } else if (tid >= 32 && tid < 32 + WW) {
        int k = tid - 32;
        float a = 0.f;
        for (int c = 0; c < CCH; ++c) a += cw[c] * sumW[c * WW + k];
        sW[k] = fmaxf(a, 0.f);
    }
    __syncthreads();

    _Float16* zb = zg + (size_t)b * SPAT;
    for (int t = tid; t < SPAT; t += 256) {
        int w = t % WW;
        int h = (t / WW) % HH;
        int d = (t / (WW * HH)) % DD;
        int c = t / (DD * HH * WW);
        float f = fw[c];
        float gd = 1.f / (1.f + __expf(-f * sH[d]));
        float gh = 1.f / (1.f + __expf(-f * sH[h]));
        float gw = 1.f / (1.f + __expf(-f * sW[w]));
        zb[t] = (_Float16)(xb[t] * gd * gh * gw);
    }
}

// ---------------------------------------------------------------------------
// WMMA f16 GEMM: out[M,N] = act(A[M,K] @ W[N,K]^T + bias)
// Block tile 128x256, BK=32, 256 threads = 8 waves (2 M-waves x 4 N-waves),
// each wave computes 64x64 via 4x4 grid of v_wmma_f32_16x16x32_f16.
// Tiles staged with double-buffered GLOBAL_LOAD_ASYNC_TO_LDS_B128 (ASYNCcnt).
// ---------------------------------------------------------------------------
__device__ __forceinline__ v16h cat8(v8h lo, v8h hi) {
    v16h r;
#pragma unroll
    for (int i = 0; i < 8; ++i) { r[i] = lo[i]; r[i + 8] = hi[i]; }
    return r;
}

// Issue one tile's async loads: A strip 32B/thread, B strip 64B/thread.
__device__ __forceinline__ void issue_tile(uint32_t asBase, uint32_t bsBase,
                                           const _Float16* __restrict__ A,
                                           const _Float16* __restrict__ W,
                                           int m0, int n0, int K, int kk, int tid)
{
    {   // A tile: 128 rows x 32 f16 (8 KB). thread -> (row = tid>>1, 32B half-row)
        const int row = tid >> 1;
        const int sel = (tid & 1) * 16;            // f16 elems
        const _Float16* g = A + (size_t)(m0 + row) * K + kk + sel;
        const uint32_t l = asBase + (uint32_t)(row * 32 + sel) * 2u;
        ASYNC_B128(l, g, 0);
        ASYNC_B128(l, g, 16);
    }
    {   // B tile: 256 rows(n) x 32 f16 (16 KB). thread -> full 64B row
        const _Float16* g = W + (size_t)(n0 + tid) * K + kk;
        const uint32_t l = bsBase + (uint32_t)tid * 64u;
        ASYNC_B128(l, g, 0);
        ASYNC_B128(l, g, 16);
        ASYNC_B128(l, g, 32);
        ASYNC_B128(l, g, 48);
    }
}

template<bool RELU, typename OUT_T>
__global__ __launch_bounds__(256)
void gemm_wmma(const _Float16* __restrict__ A,   // M x K row-major
               const _Float16* __restrict__ W,   // N x K row-major
               const float* __restrict__ bias,   // N
               OUT_T* __restrict__ out,          // M x N row-major
               int M, int N, int K)
{
    __shared__ _Float16 As[2][128 * 32];   // 2 x 8 KB
    __shared__ _Float16 Bs[2][256 * 32];   // 2 x 16 KB

    const int tid   = threadIdx.x;
    const int lane  = tid & 31;
    const int wave  = tid >> 5;
    const int waveM = wave >> 2;        // 0..1
    const int waveN = wave & 3;         // 0..3
    const int half  = lane >> 4;        // 0/1 (lane half per WMMA layout)
    const int l16   = lane & 15;

    const int m0 = blockIdx.x * 128;
    const int n0 = blockIdx.y * 256;

    const uint32_t asB[2] = { (uint32_t)(uintptr_t)&As[0][0],
                              (uint32_t)(uintptr_t)&As[1][0] };
    const uint32_t bsB[2] = { (uint32_t)(uintptr_t)&Bs[0][0],
                              (uint32_t)(uintptr_t)&Bs[1][0] };

    const v8f vzero = {0.f,0.f,0.f,0.f,0.f,0.f,0.f,0.f};
    v8f acc[4][4];
#pragma unroll
    for (int i = 0; i < 4; ++i)
#pragma unroll
        for (int j = 0; j < 4; ++j) acc[i][j] = vzero;

    const int ksteps = K >> 5;

    // prologue: stage tile 0 into buffer 0
    issue_tile(asB[0], bsB[0], A, W, m0, n0, K, 0, tid);

    for (int kt = 0; kt < ksteps; ++kt) {
        const int cur = kt & 1;
        if (kt + 1 < ksteps) {
            issue_tile(asB[cur ^ 1], bsB[cur ^ 1], A, W, m0, n0, K,
                       (kt + 1) << 5, tid);
            WAIT_ASYNCCNT(6);   // in-order: current tile's 6 loads have landed
        } else {
            WAIT_ASYNCCNT(0);
        }
        __syncthreads();        // all waves' async data visible

        const _Float16* __restrict__ as = As[cur];
        const _Float16* __restrict__ bs = Bs[cur];

        // B fragments: lane l16 = column n, half selects K group 0..15 / 16..31
        v16h bf[4];
#pragma unroll
        for (int j = 0; j < 4; ++j) {
            const _Float16* bp = bs + (waveN * 64 + j * 16 + l16) * 32 + half * 16;
            bf[j] = cat8(*(const v8h*)bp, *(const v8h*)(bp + 8));
        }
#pragma unroll
        for (int i = 0; i < 4; ++i) {
            // A fragment: lane l16 = row m; lo = K[half*16..+7], hi = K[half*16+8..+7]
            const _Float16* ap = as + (waveM * 64 + i * 16 + l16) * 32 + half * 16;
            v16h af = cat8(*(const v8h*)ap, *(const v8h*)(ap + 8));
#pragma unroll
            for (int j = 0; j < 4; ++j) {
                acc[i][j] = __builtin_amdgcn_wmma_f32_16x16x32_f16(
                    false, af, false, bf[j], (short)0, acc[i][j], false, false);
            }
        }
        __syncthreads();        // everyone done reading before buffer reuse
    }

    // epilogue: bias + optional relu; D layout: VGPR r -> row r (+8 for hi half)
#pragma unroll
    for (int i = 0; i < 4; ++i) {
#pragma unroll
        for (int j = 0; j < 4; ++j) {
            const int col = n0 + waveN * 64 + j * 16 + l16;
            const float bv = bias[col];
#pragma unroll
            for (int r = 0; r < 8; ++r) {
                const int row = m0 + waveM * 64 + i * 16 + r + half * 8;
                float v = acc[i][j][r] + bv;
                if (RELU) v = fmaxf(v, 0.f);
                out[(size_t)row * N + col] = (OUT_T)v;
            }
        }
    }
}

// ---------------------------------------------------------------------------
// Final tiny GEMM (512 -> 3) + bias + row L2-normalize. One wave32 per row.
// ---------------------------------------------------------------------------
__global__ __launch_bounds__(256)
void gemm3_norm(const float* __restrict__ z2,   // 4096 x 512
                const float* __restrict__ w3,   // 3 x 512
                const float* __restrict__ b3,   // 3
                float* __restrict__ out)        // 4096 x 3
{
    const int lane = threadIdx.x & 31;
    const int wave = threadIdx.x >> 5;
    const int row  = blockIdx.x * 8 + wave;
    const float* zr = z2 + (size_t)row * 512;
    float a0 = 0.f, a1 = 0.f, a2 = 0.f;
    for (int k = lane; k < 512; k += 32) {
        float v = zr[k];
        a0 += v * w3[k];
        a1 += v * w3[512 + k];
        a2 += v * w3[1024 + k];
    }
#pragma unroll
    for (int off = 16; off > 0; off >>= 1) {
        a0 += __shfl_xor(a0, off, 32);
        a1 += __shfl_xor(a1, off, 32);
        a2 += __shfl_xor(a2, off, 32);
    }
    if (lane == 0) {
        a0 += b3[0]; a1 += b3[1]; a2 += b3[2];
        float n = sqrtf(a0 * a0 + a1 * a1 + a2 * a2);
        float inv = 1.f / fmaxf(n, 1e-12f);
        out[row * 3 + 0] = a0 * inv;
        out[row * 3 + 1] = a1 * inv;
        out[row * 3 + 2] = a2 * inv;
    }
}

// ---------------------------------------------------------------------------
extern "C" void kernel_launch(void* const* d_in, const int* in_sizes, int n_in,
                              void* d_out, int out_size, void* d_ws, size_t ws_size,
                              hipStream_t stream) {
    const float* x      = (const float*)d_in[0];
    const float* conv_w = (const float*)d_in[1];
    const float* F_w    = (const float*)d_in[2];
    const float* w1     = (const float*)d_in[3];
    const float* b1     = (const float*)d_in[4];
    const float* w2     = (const float*)d_in[5];
    const float* b2     = (const float*)d_in[6];
    const float* w3     = (const float*)d_in[7];
    const float* b3     = (const float*)d_in[8];
    float* out = (float*)d_out;

    char* ws = (char*)d_ws;
    size_t off = 0;
    _Float16* zg  = (_Float16*)(ws + off); off += (size_t)BATCH * SPAT * 2; // 98.3 MB
    _Float16* w1h = (_Float16*)(ws + off); off += (size_t)N1 * SPAT * 2;    // 98.3 MB
    _Float16* z1  = (_Float16*)(ws + off); off += (size_t)BATCH * N1 * 2;   // 33.6 MB
    _Float16* w2h = (_Float16*)(ws + off); off += (size_t)N2 * N1 * 2;      //  4.2 MB
    float*    z2  = (float*)(ws + off);    off += (size_t)BATCH * N2 * 4;   //  8.4 MB

    // 1) weight conversions (fp32 -> f16)
    f32_to_f16<<<4096, 256, 0, stream>>>(w1, w1h, N1 * SPAT);
    f32_to_f16<<<512, 256, 0, stream>>>(w2, w2h, N2 * N1);

    // 2) gating: z = x * gd * gh * gw  (f16)
    gate_kernel<<<BATCH, 256, 0, stream>>>(x, conv_w, F_w, zg);

    // 3) z1 = relu(z @ w1^T + b1)   [4096 x 4096]
    gemm_wmma<true, _Float16><<<dim3(32, 16), 256, 0, stream>>>(
        zg, w1h, b1, z1, BATCH, N1, SPAT);

    // 4) z2 = relu(z1 @ w2^T + b2)  [4096 x 512]
    gemm_wmma<true, float><<<dim3(32, 2), 256, 0, stream>>>(
        z1, w2h, b2, z2, BATCH, N2, N1);

    // 5) out = normalize(z2 @ w3^T + b3)  [4096 x 3]
    gemm3_norm<<<512, 256, 0, stream>>>(z2, w3, b3, out);
}
